// HetNetGNN_58093727646066
// MI455X (gfx1250) — matrix operand order repaired and checked by reading, not compiled, not run on hardware
//
#include <hip/hip_runtime.h>
#include <hip/hip_bf16.h>
#include <math.h>

// ---------------------------------------------------------------------------
// HGT forward for MI455X (gfx1250, wave32, WMMA).
// Dense 64-wide GEMMs use v_wmma_f32_16x16x32_bf16 with weights pre-packed
// into WMMA B-fragment order (2x global_load_b128 per fragment).
// Relation per-head (16x16) transforms are folded into K/V projection weights.
// ---------------------------------------------------------------------------

typedef __bf16 bf16;
typedef __attribute__((ext_vector_type(16))) __bf16 v16bf;
typedef __attribute__((ext_vector_type(8)))  __bf16 v8bf;
typedef __attribute__((ext_vector_type(8)))  float  v8f;

static constexpr int kNUser = 32768;
static constexpr int kNAp   = 4096;
static constexpr int kE     = 524288;
static constexpr int kHid   = 64;
static constexpr int kHeads = 4;
static constexpr int kDh    = 16;

// ---------------------------- small utility kernels ------------------------

__global__ void fill_f32(float* p, float v, int n) {
  int i = blockIdx.x * blockDim.x + threadIdx.x;
  if (i < n) p[i] = v;
}

__global__ void f32_to_bf16(const float* __restrict__ s, bf16* __restrict__ d, int n) {
  int i = blockIdx.x * blockDim.x + threadIdx.x;
  if (i < n) d[i] = (bf16)s[i];
}

// gb[i] = bf16( gelu_tanh(agg[i]) )
__global__ void gelu_to_bf16(const float* __restrict__ a, bf16* __restrict__ g, int n) {
  int i = blockIdx.x * blockDim.x + threadIdx.x;
  if (i < n) {
    float x = a[i];
    float t = 0.7978845608028654f * (x + 0.044715f * x * x * x);
    g[i] = (bf16)(0.5f * x * (1.0f + tanhf(t)));
  }
}

// -------------------- weight packing into B-fragment order -----------------
// Packed layout: element index ((t*2 + s)*32 + lane)*16 + i holds
//   W[(32*s + (lane>>4)*16 + i) * 64 + (16*t + (lane&15))]
// so a lane's 16 bf16 B-fragment values are 32 contiguous bytes.

__global__ void pack_w(const float* __restrict__ W, bf16* __restrict__ Wp) {
  int idx = blockIdx.x * blockDim.x + threadIdx.x;  // 4096
  if (idx >= kHid * kHid) return;
  int i = idx & 15, l = (idx >> 4) & 31, s = (idx >> 9) & 1, t = idx >> 10;
  int k = 32 * s + (l >> 4) * 16 + i;
  int n = 16 * t + (l & 15);
  Wp[idx] = (bf16)W[k * kHid + n];
}

// Fold per-head relation matrix R[h][d][e] into W and b, writing W packed:
// effective Wc[k, h*16+e] = sum_d W[k, h*16+d] * R[h,d,e]
__global__ void pack_combine_wb(const float* __restrict__ W, const float* __restrict__ b,
                                const float* __restrict__ R,
                                bf16* __restrict__ Wp, float* __restrict__ bc) {
  int idx = blockIdx.x * blockDim.x + threadIdx.x;  // 4096
  if (idx >= kHid * kHid) return;
  int i = idx & 15, l = (idx >> 4) & 31, s = (idx >> 9) & 1, t = idx >> 10;
  int k = 32 * s + (l >> 4) * 16 + i;     // input row of W
  int h = t, e = l & 15;                  // output column = 16*t + (l&15)
  const float* Rh = R + h * (kDh * kDh) + e;
  float acc = 0.f;
#pragma unroll
  for (int d = 0; d < kDh; ++d) acc += W[k * kHid + h * kDh + d] * Rh[d * kDh];
  Wp[idx] = (bf16)acc;
  if (idx < kHid) {  // also produce combined bias, one thread per column
    int j = idx, hh = j >> 4, ee = j & 15;
    const float* Rb = R + hh * (kDh * kDh) + ee;
    float ba = 0.f;
#pragma unroll
    for (int d = 0; d < kDh; ++d) ba += b[hh * kDh + d] * Rb[d * kDh];
    bc[j] = ba;
  }
}

// h = relu(x[M,16] @ W[16,64] + b)
__global__ void in_proj(const float* __restrict__ x, const float* __restrict__ W,
                        const float* __restrict__ b, float* __restrict__ h, int M) {
  int idx = blockIdx.x * blockDim.x + threadIdx.x;
  if (idx >= M * kHid) return;
  int n = idx >> 6, j = idx & 63;
  float acc = b[j];
#pragma unroll
  for (int d = 0; d < 16; ++d) acc += x[n * 16 + d] * W[d * kHid + j];
  h[idx] = acc > 0.f ? acc : 0.f;
}

// out[n] = h[n,:] @ Wout[64,1] + bout
__global__ void out_proj(const float* __restrict__ h, const float* __restrict__ W,
                         const float* __restrict__ b, float* __restrict__ o, int M) {
  int n = blockIdx.x * blockDim.x + threadIdx.x;
  if (n >= M) return;
  float acc = b[0];
#pragma unroll
  for (int j = 0; j < kHid; ++j) acc += h[n * kHid + j] * W[j];
  o[n] = acc;
}

// ------------------------------- WMMA GEMMs --------------------------------
// A fragment (16-bit A 16x32, ISA 7.12.2): per lane, elements are two
// contiguous 8-element runs -> two 16B vector loads + shuffle.
__device__ __forceinline__ v16bf load_a_frag(const bf16* __restrict__ A, int row0,
                                             int lane, int kbase) {
  const bf16* Ar = A + (row0 + (lane & 15)) * kHid + kbase + (lane >> 4) * 16;
  v8bf lo = *(const v8bf*)Ar;
  v8bf hi = *(const v8bf*)(Ar + 8);
  return __builtin_shufflevector(lo, hi, 0, 1, 2, 3, 4, 5, 6, 7,
                                 8, 9, 10, 11, 12, 13, 14, 15);
}

// B fragment from packed weights: one 32B load per (tile, kstep)
__device__ __forceinline__ v16bf load_b_packed(const v16bf* __restrict__ Wp,
                                               int t, int s, int lane) {
  return Wp[(t * 2 + s) * 32 + lane];
}

__device__ __forceinline__ v8f wmma_k64(v16bf a0, v16bf a1, const v16bf* Wp,
                                        int t, int lane) {
  v16bf b0 = load_b_packed(Wp, t, 0, lane);
  v16bf b1 = load_b_packed(Wp, t, 1, lane);
  v8f c = {};
  c = __builtin_amdgcn_wmma_f32_16x16x32_bf16(false, a0, false, b0, (short)0, c, false, false);
  c = __builtin_amdgcn_wmma_f32_16x16x32_bf16(false, a1, false, b1, (short)0, c, false, false);
  return c;
}

// Fused q/kr/vr: 3 GEMMs sharing the A operand (24 WMMAs per wave).
// Outer N-tile loop kept rolled (#pragma unroll 1) to bound live accumulators
// and avoid VGPR spills.
__global__ void gemm64x3(const bf16* __restrict__ A,
                         const v16bf* __restrict__ W0, const v16bf* __restrict__ W1,
                         const v16bf* __restrict__ W2,
                         const float* __restrict__ b0, const float* __restrict__ b1,
                         const float* __restrict__ b2,
                         float* __restrict__ o0, float* __restrict__ o1,
                         float* __restrict__ o2, int M) {
  int lane = threadIdx.x & 31;
  int wave = threadIdx.x >> 5;
  int row0 = (blockIdx.x * 4 + wave) * 16;
  if (row0 >= M) return;
  v16bf a0 = load_a_frag(A, row0, lane, 0);
  v16bf a1 = load_a_frag(A, row0, lane, 32);
  int nlo = lane & 15, mhi = (lane >> 4) * 8;
  const v16bf* Ws[3] = {W0, W1, W2};
  const float* bs[3] = {b0, b1, b2};
  float* os[3] = {o0, o1, o2};
#pragma unroll 1
  for (int t = 0; t < 4; ++t) {
    int n = 16 * t + nlo;
#pragma unroll
    for (int w = 0; w < 3; ++w) {
      v8f c = wmma_k64(a0, a1, Ws[w], t, lane);
      float bn = bs[w][n];
      float* op = os[w];
#pragma unroll
      for (int r = 0; r < 8; ++r) {  // D: lanes<16 M=r, lanes>=16 M=r+8
        op[(row0 + mhi + r) * kHid + n] = c[r] + bn;
      }
    }
  }
}

// h = sigmoid(skip)*(G @ Wa + ba) + (1-sigmoid(skip))*h
__global__ void gemm64_update(const bf16* __restrict__ G, const v16bf* __restrict__ Wp,
                              const float* __restrict__ bias, const float* __restrict__ skip,
                              float* __restrict__ h, int M) {
  int lane = threadIdx.x & 31;
  int wave = threadIdx.x >> 5;
  int row0 = (blockIdx.x * 4 + wave) * 16;
  if (row0 >= M) return;
  float gate = 1.f / (1.f + __expf(-skip[0]));
  v16bf a0 = load_a_frag(G, row0, lane, 0);
  v16bf a1 = load_a_frag(G, row0, lane, 32);
  int nlo = lane & 15, mhi = (lane >> 4) * 8;
#pragma unroll 1
  for (int t = 0; t < 4; ++t) {
    v8f c = wmma_k64(a0, a1, Wp, t, lane);
    int n = 16 * t + nlo;
    float bn = bias[n];
#pragma unroll
    for (int r = 0; r < 8; ++r) {
      int idx = (row0 + mhi + r) * kHid + n;
      h[idx] = gate * (c[r] + bn) + (1.f - gate) * h[idx];
    }
  }
}

// ----------------------------- edge kernels --------------------------------

__device__ __forceinline__ void atomicMaxFloat(float* addr, float val) {
  if (val >= 0.f) atomicMax((int*)addr, __float_as_int(val));
  else            atomicMin((unsigned int*)addr, __float_as_uint(val));
}

// pass 1: logits[e,h] = <q[dst], kr[src]>_h * p[h]/sqrt(DH); atomic max per (dst,h)
__global__ void edge_logits_max(const float* __restrict__ q, const float* __restrict__ kr,
                                const int* __restrict__ si, const int* __restrict__ di,
                                const float* __restrict__ p, float* __restrict__ logits,
                                float* __restrict__ mbuf, int E) {
  int e = blockIdx.x * blockDim.x + threadIdx.x;
  if (e >= E) return;
  int d = di[e];
  const float* qd = q + d * kHid;
  const float* ks = kr + si[e] * kHid;
#pragma unroll
  for (int h = 0; h < kHeads; ++h) {
    float acc = 0.f;
#pragma unroll
    for (int t = 0; t < kDh; ++t) acc += qd[h * kDh + t] * ks[h * kDh + t];
    float l = acc * p[h] * 0.25f;  // /sqrt(16)
    logits[e * kHeads + h] = l;
    atomicMaxFloat(mbuf + d * kHeads + h, l);
  }
}

// pass 2: ev = exp(l - m); atomic sum; store ev in place of logits
__global__ void edge_expsum(const int* __restrict__ di, const float* __restrict__ mbuf,
                            float* __restrict__ logits, float* __restrict__ sbuf, int E) {
  int e = blockIdx.x * blockDim.x + threadIdx.x;
  if (e >= E) return;
  int d = di[e];
#pragma unroll
  for (int h = 0; h < kHeads; ++h) {
    float mm = mbuf[d * kHeads + h];
    if (!__builtin_isfinite(mm)) mm = 0.f;
    float ev = __expf(logits[e * kHeads + h] - mm);
    logits[e * kHeads + h] = ev;
    atomicAdd(sbuf + d * kHeads + h, ev);
  }
}

// pass 3: agg[dst,h,:] += (ev/s) * vr[src,h,:]   (thread = (edge, head))
__global__ void edge_scatter(const int* __restrict__ si, const int* __restrict__ di,
                             const float* __restrict__ ev, const float* __restrict__ sbuf,
                             const float* __restrict__ vr, float* __restrict__ agg, int E) {
  int t = blockIdx.x * blockDim.x + threadIdx.x;
  if (t >= E * kHeads) return;
  int e = t >> 2, h = t & 3;
  int s = si[e], d = di[e];
  float alpha = ev[e * kHeads + h] / (sbuf[d * kHeads + h] + 1e-16f);
  const float* vs = vr + s * kHid + h * kDh;
  float* ad = agg + d * kHid + h * kDh;
#pragma unroll
  for (int k = 0; k < kDh; ++k) atomicAdd(ad + k, alpha * vs[k]);
}

// ------------------------------ host driver --------------------------------
//
// Input flattening (top-level dict in insertion order; params pytree with
// sorted dict keys, lists/tuples in order):
//  0 x_user  1 x_ap
//  2 in.ap.W  3 in.ap.b  4 in.user.W  5 in.user.b
//  per layer l (base = 6 + 24*l):
//   +0 ap.a.W +1 ap.a.b +2 ap.k.W +3 ap.k.b +4 ap.q.W +5 ap.q.b +6 ap.skip
//   +7 ap.v.W +8 ap.v.b
//   +9 rel.a2u.a +10 rel.a2u.m +11 rel.a2u.p
//   +12 rel.u2a.a +13 rel.u2a.m +14 rel.u2a.p
//   +15 user.a.W +16 user.a.b +17 user.k.W +18 user.k.b +19 user.q.W
//   +20 user.q.b +21 user.skip +22 user.v.W +23 user.v.b
//  54 out.W  55 out.b  56 edge_u2a  57 edge_a2u

extern "C" void kernel_launch(void* const* d_in, const int* in_sizes, int n_in,
                              void* d_out, int out_size, void* d_ws, size_t ws_size,
                              hipStream_t stream) {
  (void)in_sizes; (void)n_in; (void)out_size; (void)ws_size;
  auto F = [&](int i) { return (const float*)d_in[i]; };
  const float* x_user = F(0);
  const float* x_ap   = F(1);
  const int* edge_u2a = (const int*)d_in[56];
  const int* edge_a2u = (const int*)d_in[57];

  // workspace bump allocator (256B aligned)
  char* ws = (char*)d_ws;
  size_t off = 0;
  auto alloc = [&](size_t bytes) -> void* {
    off = (off + 255) & ~(size_t)255;
    void* p = ws + off;
    off += bytes;
    return p;
  };

  float* h_u   = (float*)alloc(sizeof(float) * kNUser * kHid);
  float* h_a   = (float*)alloc(sizeof(float) * kNAp * kHid);
  bf16*  hb_u  = (bf16*)alloc(sizeof(bf16) * kNUser * kHid);
  bf16*  hb_a  = (bf16*)alloc(sizeof(bf16) * kNAp * kHid);
  float* q_u   = (float*)alloc(sizeof(float) * kNUser * kHid);
  float* kr_u  = (float*)alloc(sizeof(float) * kNUser * kHid);
  float* vr_u  = (float*)alloc(sizeof(float) * kNUser * kHid);
  float* q_a   = (float*)alloc(sizeof(float) * kNAp * kHid);
  float* kr_a  = (float*)alloc(sizeof(float) * kNAp * kHid);
  float* vr_a  = (float*)alloc(sizeof(float) * kNAp * kHid);
  float* agg_u = (float*)alloc(sizeof(float) * kNUser * kHid);
  float* agg_a = (float*)alloc(sizeof(float) * kNAp * kHid);
  bf16*  gb_u  = (bf16*)alloc(sizeof(bf16) * kNUser * kHid);
  bf16*  gb_a  = (bf16*)alloc(sizeof(bf16) * kNAp * kHid);
  float* m_u   = (float*)alloc(sizeof(float) * kNUser * kHeads);
  float* s_u   = (float*)alloc(sizeof(float) * kNUser * kHeads);
  float* m_a   = (float*)alloc(sizeof(float) * kNAp * kHeads);
  float* s_a   = (float*)alloc(sizeof(float) * kNAp * kHeads);
  float* lg0   = (float*)alloc(sizeof(float) * kE * kHeads);  // u2a
  float* lg1   = (float*)alloc(sizeof(float) * kE * kHeads);  // a2u
  bf16*  wq_p  = (bf16*)alloc(sizeof(bf16) * kHid * kHid);
  bf16*  wk_p  = (bf16*)alloc(sizeof(bf16) * kHid * kHid);
  bf16*  wv_p  = (bf16*)alloc(sizeof(bf16) * kHid * kHid);
  bf16*  wa_p  = (bf16*)alloc(sizeof(bf16) * kHid * kHid);
  float* bk_c  = (float*)alloc(sizeof(float) * kHid);
  float* bv_c  = (float*)alloc(sizeof(float) * kHid);

  auto cdiv = [](int a, int b) { return (a + b - 1) / b; };

  // input projections
  in_proj<<<cdiv(kNUser * kHid, 256), 256, 0, stream>>>(x_user, F(4), F(5), h_u, kNUser);
  in_proj<<<cdiv(kNAp * kHid, 256), 256, 0, stream>>>(x_ap, F(2), F(3), h_a, kNAp);

  struct NT {
    const float *aW, *ab, *kW, *kb, *qW, *qb, *skip, *vW, *vb, *relA, *relM;
    float *h, *q, *kr, *vr, *agg, *m, *s;
    bf16 *hb, *gb;
    int n;
  };

  for (int l = 0; l < 2; ++l) {
    int B = 6 + 24 * l;
    NT ap  = {F(B + 0), F(B + 1), F(B + 2), F(B + 3), F(B + 4), F(B + 5), F(B + 6),
              F(B + 7), F(B + 8), F(B + 9), F(B + 10),  // ap is src of a2u
              h_a, q_a, kr_a, vr_a, agg_a, m_a, s_a, hb_a, gb_a, kNAp};
    NT usr = {F(B + 15), F(B + 16), F(B + 17), F(B + 18), F(B + 19), F(B + 20), F(B + 21),
              F(B + 22), F(B + 23), F(B + 12), F(B + 13),  // user is src of u2a
              h_u, q_u, kr_u, vr_u, agg_u, m_u, s_u, hb_u, gb_u, kNUser};
    const float* p_u2a = F(B + 14);
    const float* p_a2u = F(B + 11);

    NT* nts[2] = {&usr, &ap};
    for (int i = 0; i < 2; ++i) {
      NT* t = nts[i];
      int n64 = t->n * kHid;
      f32_to_bf16<<<cdiv(n64, 256), 256, 0, stream>>>(t->h, t->hb, n64);
      pack_w<<<cdiv(kHid * kHid, 256), 256, 0, stream>>>(t->qW, wq_p);
      pack_combine_wb<<<cdiv(kHid * kHid, 256), 256, 0, stream>>>(t->kW, t->kb, t->relA,
                                                                  wk_p, bk_c);
      pack_combine_wb<<<cdiv(kHid * kHid, 256), 256, 0, stream>>>(t->vW, t->vb, t->relM,
                                                                  wv_p, bv_c);
      gemm64x3<<<t->n / 64, 128, 0, stream>>>(t->hb, (const v16bf*)wq_p, (const v16bf*)wk_p,
                                              (const v16bf*)wv_p, t->qb, bk_c, bv_c,
                                              t->q, t->kr, t->vr, t->n);
      fill_f32<<<cdiv(t->n * kHeads, 256), 256, 0, stream>>>(t->m, -INFINITY, t->n * kHeads);
      fill_f32<<<cdiv(t->n * kHeads, 256), 256, 0, stream>>>(t->s, 0.f, t->n * kHeads);
      fill_f32<<<cdiv(n64, 256), 256, 0, stream>>>(t->agg, 0.f, n64);
    }

    // relation u2a: src=user, dst=ap ; relation a2u: src=ap, dst=user
    const int* si0 = edge_u2a;          const int* di0 = edge_u2a + kE;
    const int* si1 = edge_a2u;          const int* di1 = edge_a2u + kE;
    int gE = cdiv(kE, 256), gEH = cdiv(kE * kHeads, 256);
    edge_logits_max<<<gE, 256, 0, stream>>>(q_a, kr_u, si0, di0, p_u2a, lg0, m_a, kE);
    edge_logits_max<<<gE, 256, 0, stream>>>(q_u, kr_a, si1, di1, p_a2u, lg1, m_u, kE);
    edge_expsum<<<gE, 256, 0, stream>>>(di0, m_a, lg0, s_a, kE);
    edge_expsum<<<gE, 256, 0, stream>>>(di1, m_u, lg1, s_u, kE);
    edge_scatter<<<gEH, 256, 0, stream>>>(si0, di0, lg0, s_a, vr_u, agg_a, kE);
    edge_scatter<<<gEH, 256, 0, stream>>>(si1, di1, lg1, s_u, vr_a, agg_u, kE);

    for (int i = 0; i < 2; ++i) {
      NT* t = nts[i];
      int n64 = t->n * kHid;
      gelu_to_bf16<<<cdiv(n64, 256), 256, 0, stream>>>(t->agg, t->gb, n64);
      pack_w<<<cdiv(kHid * kHid, 256), 256, 0, stream>>>(t->aW, wa_p);
      gemm64_update<<<t->n / 64, 128, 0, stream>>>(t->gb, (const v16bf*)wa_p, t->ab,
                                                   t->skip, t->h, t->n);
    }
  }

  out_proj<<<cdiv(kNUser, 256), 256, 0, stream>>>(h_u, F(54), F(55), (float*)d_out, kNUser);
}
